// Model_45681272160674
// MI455X (gfx1250) — compile-verified, matrix operand order checked by reading, not run on hardware
//
#include <hip/hip_runtime.h>
#include <hip/hip_bf16.h>
#include <math.h>

#define HID     512
#define BATCH   256
#define TSTEPS  784
#define NCLS    10

typedef __bf16 bf16_t;
typedef __attribute__((ext_vector_type(16))) __bf16 v16bf;
typedef __attribute__((ext_vector_type(8)))  __bf16 v8bf;
typedef __attribute__((ext_vector_type(8)))  float  v8f;

// ---------------------------------------------------------------------------
// Prep: convert W_hh (f32, row-major HIDxHID) to bf16 in ws; zero h0 (bf16).
// ---------------------------------------------------------------------------
__global__ void __launch_bounds__(256)
prep_kernel(const float* __restrict__ Whh,
            bf16_t* __restrict__ Whh_bf,
            bf16_t* __restrict__ h0) {
  int tid = blockIdx.x * blockDim.x + threadIdx.x;
  if (tid < HID * HID) Whh_bf[tid] = (bf16_t)Whh[tid];
  if (tid < BATCH * HID) h0[tid] = (bf16_t)0.0f;
}

// ---------------------------------------------------------------------------
// One recurrence step:  h_next = tanh( x_t * w_ih + h_prev @ W_hh^T + b )
//
// Grid:  (BATCH/16, HID/128), block = 256 threads = 8 waves (wave32).
// Block covers a 16(batch) x 128(hid) output tile; each wave owns one
// 16x16 tile and sweeps K = 0..512 with v_wmma_f32_16x16x32_bf16.
//
// A (16x32 bf16)  = h_prev[b0:b0+16, k0:k0+32]     (staged in LDS, shared)
// B (32x16 bf16)  : B[K][N] = W_hh[i0+N][k0+K]     (row-of-W_hh reads, L2)
// D (16x16 f32)   : fused epilogue adds x*w_ih + b, tanh, bf16 store.
// ---------------------------------------------------------------------------
__global__ void __launch_bounds__(256)
rnn_step_kernel(const bf16_t* __restrict__ hprev,
                bf16_t* __restrict__ hnext,
                const bf16_t* __restrict__ Whh_bf,
                const float* __restrict__ inputs,
                const int* __restrict__ order,
                const float* __restrict__ w_ih,
                const float* __restrict__ bias,
                int t) {
  __shared__ __align__(16) bf16_t sh[16 * HID];   // 16 KB h_prev tile
  __shared__ float xsh[16];

  const int tid  = threadIdx.x;
  const int b0   = blockIdx.x * 16;
  const int wave = tid >> 5;
  const int lane = tid & 31;
  const int i0   = blockIdx.y * 128 + wave * 16;

  // Cooperative stage of h_prev[b0:b0+16, :] into LDS as 16B chunks.
  {
    const uint4* src = (const uint4*)(hprev + b0 * HID);
    uint4*       dst = (uint4*)sh;
    #pragma unroll
    for (int k = 0; k < (16 * HID / 8) / 256; ++k)   // 4 iterations
      dst[tid + k * 256] = src[tid + k * 256];
  }
  if (tid < 16) {
    int ts = order[t];
    xsh[tid] = inputs[(b0 + tid) * TSTEPS + ts];
  }
  __syncthreads();

  const int m   = lane & 15;       // A: row M,  B/D: col N, all = lane&15
  const int hi  = lane >> 4;
  const int off = hi * 8;          // K sub-offset for upper half-wave
  const int ncol = i0 + m;         // global hid column this lane produces

  const bf16_t* brow = Whh_bf + ncol * HID;
  // CDNA5 prefetch of this wave's weight rows (global_prefetch_b8)
  __builtin_prefetch(brow, 0, 0);

  v8f acc = {};
  #pragma unroll 4
  for (int k0 = 0; k0 < HID; k0 += 32) {
    union { v16bf v; v8bf h2[2]; } A, B;
    // A fragment: lane holds M = lane&15; K chunks [k0+off, +7], [k0+16+off, +7]
    A.h2[0] = *(const v8bf*)(sh + m * HID + k0 + off);
    A.h2[1] = *(const v8bf*)(sh + m * HID + k0 + 16 + off);
    // B fragment: lane holds N = lane&15; same K chunk pattern along W_hh row
    B.h2[0] = *(const v8bf*)(brow + k0 + off);
    B.h2[1] = *(const v8bf*)(brow + k0 + 16 + off);
    acc = __builtin_amdgcn_wmma_f32_16x16x32_bf16(
        /*neg_a=*/false, A.v, /*neg_b=*/false, B.v,
        /*c_mod=*/(short)0, acc, /*reuse_a=*/false, /*reuse_b=*/false);
  }

  // Fused epilogue: acc[r] is D[M = r + 8*hi][N = lane&15]
  const float wih_c  = w_ih[ncol];
  const float bias_c = bias[ncol];
  #pragma unroll
  for (int r = 0; r < 8; ++r) {
    int row = r + hi * 8;                       // local batch row
    float v = acc[r] + xsh[row] * wih_c + bias_c;
    hnext[(b0 + row) * HID + ncol] = (bf16_t)tanhf(v);
  }
}

// ---------------------------------------------------------------------------
// Head: logits = h @ W_out^T + b_out; log_softmax; loss + correct count.
// Single block, 256 threads, one batch element per thread (trivial FLOPs).
// ---------------------------------------------------------------------------
__global__ void __launch_bounds__(256)
head_kernel(const bf16_t* __restrict__ hfin,
            const float* __restrict__ Wout,
            const float* __restrict__ bout,
            const int* __restrict__ y,
            float* __restrict__ out) {
  __shared__ float red[256];
  __shared__ int   redi[256];
  const int b = threadIdx.x;

  float logits[NCLS];
  #pragma unroll
  for (int c = 0; c < NCLS; ++c) logits[c] = bout[c];
  for (int j = 0; j < HID; ++j) {
    float hv = (float)hfin[b * HID + j];
    #pragma unroll
    for (int c = 0; c < NCLS; ++c) logits[c] += hv * Wout[c * HID + j];
  }

  float mx = logits[0];
  int pred = 0;
  #pragma unroll
  for (int c = 1; c < NCLS; ++c)
    if (logits[c] > mx) { mx = logits[c]; pred = c; }   // first-max like argmax
  float se = 0.0f;
  #pragma unroll
  for (int c = 0; c < NCLS; ++c) se += expf(logits[c] - mx);
  const int yb = y[b];
  const float logp = logits[yb] - mx - logf(se);

  red[b]  = -logp;
  redi[b] = (pred == yb) ? 1 : 0;
  __syncthreads();
  for (int s = 128; s > 0; s >>= 1) {
    if (b < s) { red[b] += red[b + s]; redi[b] += redi[b + s]; }
    __syncthreads();
  }
  if (b == 0) {
    out[0] = red[0] / (float)BATCH;   // mean NLL loss
    out[1] = (float)redi[0];          // correct count
  }
}

// ---------------------------------------------------------------------------
// kernel_launch: prep -> 784 captured step launches (ping-pong h) -> head.
// ws layout: [0, 512KB) W_hh bf16 | [512KB, 768KB) h ping | [768KB, 1MB) h pong
// ---------------------------------------------------------------------------
extern "C" void kernel_launch(void* const* d_in, const int* in_sizes, int n_in,
                              void* d_out, int out_size, void* d_ws, size_t ws_size,
                              hipStream_t stream) {
  const float* inputs = (const float*)d_in[0];
  const int*   y      = (const int*)  d_in[1];
  const int*   order  = (const int*)  d_in[2];
  const float* W_ih   = (const float*)d_in[3];
  const float* W_hh   = (const float*)d_in[4];
  const float* bvec   = (const float*)d_in[5];
  const float* W_out  = (const float*)d_in[6];
  const float* b_out  = (const float*)d_in[7];

  char* ws = (char*)d_ws;
  bf16_t* Whh_bf = (bf16_t*)(ws);
  bf16_t* hA     = (bf16_t*)(ws + 512 * 1024);
  bf16_t* hB     = (bf16_t*)(ws + 512 * 1024 + 256 * 1024);

  prep_kernel<<<(HID * HID + 255) / 256, 256, 0, stream>>>(W_hh, Whh_bf, hA);

  bf16_t* prev = hA;
  bf16_t* next = hB;
  for (int t = 0; t < TSTEPS; ++t) {
    rnn_step_kernel<<<dim3(BATCH / 16, HID / 128), 256, 0, stream>>>(
        prev, next, Whh_bf, inputs, order, W_ih, bvec, t);
    bf16_t* tmp = prev; prev = next; next = tmp;
  }

  head_kernel<<<1, 256, 0, stream>>>(prev, W_out, b_out, y, (float*)d_out);
}
